// SelfAttention_11321533792837
// MI455X (gfx1250) — compile-verified, matrix operand order checked by reading, not run on hardware
//
#include <hip/hip_runtime.h>
#include <hip/hip_bf16.h>

// ---------------------------------------------------------------------------
// CDNA5 (gfx1250) transformer block: QKV proj -> flash attention -> out proj
// + residual -> FFN(leaky_relu) + residual.  All dense math on
// v_wmma_f32_16x16x32_bf16 (fp32 accumulate); GEMM staging uses the gfx1250
// async global->LDS engine (ASYNCcnt) with double buffering.
// B=2, T=2048, D=1024, H=16, DH=64.
// ---------------------------------------------------------------------------

typedef __bf16 bf16;
typedef __attribute__((ext_vector_type(16))) bf16  v16bf;
typedef __attribute__((ext_vector_type(8)))  float v8f;
typedef __attribute__((ext_vector_type(4)))  int   v4i;

enum { EPI_NONE = 0, EPI_RES = 1, EPI_LEAKY = 2 };

static constexpr int Bc = 2, Tc = 2048, Dc = 1024, Hc = 16, DHc = 64;
static constexpr int Mrows = Bc * Tc; // 4096

// ---- gfx1250 async global->LDS copy (guarded; sync fallback) --------------
// Probe-learned signature: (v4i addrspace(1)*, v4i addrspace(3)*, Imm, Imm)
#if __has_builtin(__builtin_amdgcn_global_load_async_to_lds_b128)
#define HAVE_ASYNC_LDS 1
#define ASYNC_CP_B128(gsrc, ldst)                                            \
  __builtin_amdgcn_global_load_async_to_lds_b128(                            \
      (__attribute__((address_space(1))) v4i*)(const void*)(gsrc),           \
      (__attribute__((address_space(3))) v4i*)(void*)(ldst), 0, 0)
#if __has_builtin(__builtin_amdgcn_s_wait_asynccnt)
#define ASYNC_WAIT() __builtin_amdgcn_s_wait_asynccnt(0)
#else
#define ASYNC_WAIT() asm volatile("s_wait_asynccnt 0x0" ::: "memory")
#endif
#else
#define HAVE_ASYNC_LDS 0
#define ASYNC_CP_B128(gsrc, ldst)                                            \
  (*(uint4*)(ldst) = *(const uint4*)(gsrc))
#define ASYNC_WAIT() ((void)0)
#endif

__device__ __forceinline__ v8f zero8() {
  v8f z;
#pragma unroll
  for (int i = 0; i < 8; ++i) z[i] = 0.0f;
  return z;
}

// Load one 16x16x32 bf16 operand fragment: 8 bf16 at p, 8 bf16 at p+16.
// (CDNA5 16-bit A/B layout: lanes 0-15 take K = base+0..7 / base+16..23;
// lanes 16-31 take K = base+8..15 / base+24..31 -> caller adds hi8.)
__device__ __forceinline__ v16bf ld_frag(const bf16* p) {
  union { v16bf v; uint4 q[2]; } u;
  u.q[0] = *(const uint4*)p;
  u.q[1] = *(const uint4*)(p + 16);
  return u.v;
}

__device__ __forceinline__ v8f wmma_bf16(v16bf a, v16bf b, v8f c) {
  return __builtin_amdgcn_wmma_f32_16x16x32_bf16(
      /*neg_a=*/false, a, /*neg_b=*/false, b,
      /*c_mod=*/(short)0, c, /*reuse_a=*/false, /*reuse_b=*/false);
}

// ---------------------------------------------------------------------------
// fp32 -> bf16 copy
// ---------------------------------------------------------------------------
__global__ __launch_bounds__(256) void cvt_kernel(const float* __restrict__ in,
                                                  bf16* __restrict__ out, int n) {
  int i = blockIdx.x * blockDim.x + threadIdx.x;
  if (i < n) out[i] = (bf16)in[i];
}

// ---------------------------------------------------------------------------
// Weight transpose + convert: W[K,N] fp32 -> Wt[N,K] bf16  (K=N=1024)
// ---------------------------------------------------------------------------
__global__ __launch_bounds__(256) void wtrans_kernel(const float* __restrict__ W,
                                                     bf16* __restrict__ Wt) {
  __shared__ float tile[32][33];
  const int K = Dc, N = Dc;
  const int n = blockIdx.x * 32 + threadIdx.x;
  const int kbase = blockIdx.y * 32;
#pragma unroll
  for (int j = threadIdx.y; j < 32; j += 8)
    tile[j][threadIdx.x] = W[(size_t)(kbase + j) * N + n];
  __syncthreads();
  const int nw = blockIdx.x * 32;
#pragma unroll
  for (int j = threadIdx.y; j < 32; j += 8)
    Wt[(size_t)(nw + j) * K + kbase + threadIdx.x] = (bf16)tile[threadIdx.x][j];
}

// ---------------------------------------------------------------------------
// GEMM: C[M,N] = A[M,K](bf16) * Wt[N,K](bf16)^T + bias (+residual / leaky)
// Block tile 128x128, BK=32, 256 threads = 8 waves (2x4), wave tile 64x32.
// Double-buffered LDS; tile k+1 staged via async global->LDS while tile k
// runs on the WMMA pipes.
// ---------------------------------------------------------------------------
template <int EPI, bool WF32, bool WBF16>
__global__ __launch_bounds__(256) void gemm_bf16_kernel(
    const bf16* __restrict__ A, const bf16* __restrict__ Bt,
    const float* __restrict__ bias, const float* __restrict__ R,
    float* __restrict__ Cf, bf16* __restrict__ Cb, int M, int N, int K) {
  constexpr int BM = 128, BN = 128, BK = 32, LDK = BK + 8; // 80B rows, 16B-aligned
  __shared__ bf16 As[2][BM * LDK];
  __shared__ bf16 Bs[2][BN * LDK];

  const int tid = threadIdx.x;
  const int lane = tid & 31, wave = tid >> 5;
  const int wm = wave & 1, wn = wave >> 1; // 2 x 4 wave grid
  const int m0 = blockIdx.y * BM, n0 = blockIdx.x * BN;
  const int l15 = lane & 15, hi8 = (lane >> 4) * 8;

  // Stage one BMxBK + BNxBK tile pair: 512 B128 chunks / 256 threads.
  auto stage = [&](int kk2, int buf2) {
    const int kb = kk2 * BK;
#pragma unroll
    for (int i = 0; i < 2; ++i) {
      const int c = tid + 256 * i;           // 0..511
      const int row = c >> 2, kc = (c & 3) * 8;
      ASYNC_CP_B128(&A[(size_t)(m0 + row) * K + kb + kc],
                    &As[buf2][row * LDK + kc]);
      ASYNC_CP_B128(&Bt[(size_t)(n0 + row) * K + kb + kc],
                    &Bs[buf2][row * LDK + kc]);
    }
  };

  v8f acc[4][2];
#pragma unroll
  for (int mi = 0; mi < 4; ++mi)
#pragma unroll
    for (int ni = 0; ni < 2; ++ni) acc[mi][ni] = zero8();

  const int nstep = K / BK;
  stage(0, 0);
  for (int kk = 0; kk < nstep; ++kk) {
    const int buf = kk & 1;
    ASYNC_WAIT();        // tile kk landed in LDS (this wave's copies)
    __syncthreads();     // ...published to all waves; prev buffer free
    if (kk + 1 < nstep) stage(kk + 1, buf ^ 1); // overlap with compute

    v16bf af[4], bfr[2];
#pragma unroll
    for (int mi = 0; mi < 4; ++mi)
      af[mi] = ld_frag(&As[buf][(wm * 64 + mi * 16 + l15) * LDK + hi8]);
#pragma unroll
    for (int ni = 0; ni < 2; ++ni)
      bfr[ni] = ld_frag(&Bs[buf][(wn * 32 + ni * 16 + l15) * LDK + hi8]);
#pragma unroll
    for (int mi = 0; mi < 4; ++mi)
#pragma unroll
      for (int ni = 0; ni < 2; ++ni)
        acc[mi][ni] = wmma_bf16(af[mi], bfr[ni], acc[mi][ni]);
  }

  // Epilogue: C layout -> lanes 0-15 rows v, lanes 16-31 rows v+8; N = lane&15.
#pragma unroll
  for (int mi = 0; mi < 4; ++mi) {
#pragma unroll
    for (int ni = 0; ni < 2; ++ni) {
      const int col = n0 + wn * 32 + ni * 16 + l15;
      const float bcol = bias[col];
#pragma unroll
      for (int v = 0; v < 8; ++v) {
        const int row = m0 + wm * 64 + mi * 16 + v + hi8;
        float val = acc[mi][ni][v] + bcol;
        if (EPI == EPI_LEAKY) val = (val > 0.0f) ? val : 0.01f * val;
        const size_t off = (size_t)row * N + col;
        if (EPI == EPI_RES) val += R[off];
        if (WF32) Cf[off] = val;
        if (WBF16) Cb[off] = (bf16)val;
      }
    }
  }
}

// ---------------------------------------------------------------------------
// Flash attention: ctx[b,t,h,:] = softmax(q k^T / 8) v, streamed over keys.
// Grid: (B*H, T/64).  128 threads = 4 waves; each wave owns 16 query rows.
// 32 keys per iteration: 4 WMMAs for S, 4 WMMAs for P*V per wave.
// ---------------------------------------------------------------------------
__global__ __launch_bounds__(128) void flash_attn_kernel(
    const bf16* __restrict__ Qm, const bf16* __restrict__ Km,
    const bf16* __restrict__ Vm, bf16* __restrict__ Om) {
  constexpr int KT = 32;
  __shared__ bf16 Vt[DHc * 40];     // V tile transposed [d][key], padded rows
  __shared__ bf16 Ps[4 * 16 * 40];  // per-wave P tile (C-layout -> A-layout)

  const int tid = threadIdx.x, lane = tid & 31, wave = tid >> 5;
  const int bh = blockIdx.x, b = bh >> 4, h = bh & 15;
  const int q0 = blockIdx.y * 64 + wave * 16;
  const int l15 = lane & 15, hi8 = (lane >> 4) * 8;
  const size_t hoff = (size_t)h * DHc;

  auto rowp = [&](const bf16* base, int t) -> const bf16* {
    return base + (size_t)(b * Tc + t) * Dc + hoff;
  };

  // Q fragments (d 0..31 and 32..63), loaded once.
  v16bf aq[2];
  {
    const bf16* qr = rowp(Qm, q0 + l15);
    aq[0] = ld_frag(qr + 0 + hi8);
    aq[1] = ld_frag(qr + 32 + hi8);
  }

  float mrow[8], lrow[8];
  v8f o[4];
#pragma unroll
  for (int v = 0; v < 8; ++v) { mrow[v] = -1e30f; lrow[v] = 0.0f; }
#pragma unroll
  for (int dt = 0; dt < 4; ++dt) o[dt] = zero8();

  for (int kt = 0; kt < Tc; kt += KT) {
    __syncthreads(); // protect Vt from previous iteration's readers
    // Stage V tile transposed: 32 keys x 64 d -> Vt[d][key]
#pragma unroll
    for (int i = 0; i < 2; ++i) {
      const int c = tid + 128 * i;
      const int key = c >> 3, dc = (c & 7) * 8;
      union { uint4 q; bf16 hv[8]; } u;
      u.q = *(const uint4*)(rowp(Vm, kt + key) + dc);
#pragma unroll
      for (int j = 0; j < 8; ++j) Vt[(dc + j) * 40 + key] = u.hv[j];
    }
    __syncthreads();

    if (kt + KT < Tc) { // hint next K/V tiles toward the caches
      __builtin_prefetch(rowp(Km, kt + KT + l15), 0, 1);
      __builtin_prefetch(rowp(Vm, kt + KT + (tid >> 2)), 0, 1);
    }

    // S = Q K^T : K rows are d-major -> direct WMMA B fragments from global.
    v8f s[2];
#pragma unroll
    for (int nt = 0; nt < 2; ++nt) {
      const bf16* kr = rowp(Km, kt + nt * 16 + l15);
      v16bf bk0 = ld_frag(kr + 0 + hi8);
      v16bf bk1 = ld_frag(kr + 32 + hi8);
      v8f z = zero8();
      z = wmma_bf16(aq[0], bk0, z);
      z = wmma_bf16(aq[1], bk1, z);
      s[nt] = z;
    }

    // Online softmax update (per row; 16-lane shuffle reductions).
    float corr[8];
#pragma unroll
    for (int v = 0; v < 8; ++v) {
      float a0 = s[0][v] * 0.125f, a1 = s[1][v] * 0.125f;
      float tm = fmaxf(a0, a1);
      tm = fmaxf(tm, __shfl_xor(tm, 1));
      tm = fmaxf(tm, __shfl_xor(tm, 2));
      tm = fmaxf(tm, __shfl_xor(tm, 4));
      tm = fmaxf(tm, __shfl_xor(tm, 8));
      const float mn = fmaxf(mrow[v], tm);
      corr[v] = __expf(mrow[v] - mn);
      mrow[v] = mn;
      const float p0 = __expf(a0 - mn), p1 = __expf(a1 - mn);
      s[0][v] = p0;
      s[1][v] = p1;
      float ps = p0 + p1;
      ps += __shfl_xor(ps, 1);
      ps += __shfl_xor(ps, 2);
      ps += __shfl_xor(ps, 4);
      ps += __shfl_xor(ps, 8);
      lrow[v] = lrow[v] * corr[v] + ps;
    }
#pragma unroll
    for (int dt = 0; dt < 4; ++dt)
#pragma unroll
      for (int v = 0; v < 8; ++v) o[dt][v] *= corr[v];

    // P: C-layout -> A-layout via wave-private LDS bounce (bf16 convert).
#pragma unroll
    for (int nt = 0; nt < 2; ++nt)
#pragma unroll
      for (int v = 0; v < 8; ++v)
        Ps[wave * 640 + (v + hi8) * 40 + nt * 16 + l15] = (bf16)s[nt][v];
    const v16bf ap = ld_frag(&Ps[wave * 640 + l15 * 40 + hi8]);

    // O += P * V  (V read transposed from LDS as B fragments)
#pragma unroll
    for (int dt = 0; dt < 4; ++dt) {
      v16bf bv = ld_frag(&Vt[(dt * 16 + l15) * 40 + hi8]);
      o[dt] = wmma_bf16(ap, bv, o[dt]);
    }
  }

  // Normalize and store ctx as bf16 (feeds the out-projection GEMM).
#pragma unroll
  for (int dt = 0; dt < 4; ++dt) {
#pragma unroll
    for (int v = 0; v < 8; ++v) {
      const int qrow = q0 + v + hi8;
      const float val = o[dt][v] / lrow[v];
      Om[(size_t)(b * Tc + qrow) * Dc + hoff + dt * 16 + l15] = (bf16)val;
    }
  }
}

// ---------------------------------------------------------------------------
// Host-side orchestration
// ---------------------------------------------------------------------------
extern "C" void kernel_launch(void* const* d_in, const int* in_sizes, int n_in,
                              void* d_out, int out_size, void* d_ws,
                              size_t ws_size, hipStream_t stream) {
  (void)in_sizes; (void)n_in; (void)out_size; (void)ws_size;
  const float* x  = (const float*)d_in[0];
  const float* Wq = (const float*)d_in[1];  const float* bq = (const float*)d_in[2];
  const float* Wk = (const float*)d_in[3];  const float* bk = (const float*)d_in[4];
  const float* Wv = (const float*)d_in[5];  const float* bv = (const float*)d_in[6];
  const float* Wo = (const float*)d_in[7];  const float* bo = (const float*)d_in[8];
  const float* Wh = (const float*)d_in[9];  const float* bh = (const float*)d_in[10];
  const float* Wy = (const float*)d_in[11]; const float* by = (const float*)d_in[12];

  char* ws = (char*)d_ws;
  const size_t MB = 1024 * 1024;
  bf16*  xb   = (bf16*)(ws + 0 * MB);   // 8 MB  : x bf16
  bf16*  Wqt  = (bf16*)(ws + 8 * MB);   // 2 MB each, transposed bf16 weights
  bf16*  Wkt  = (bf16*)(ws + 10 * MB);
  bf16*  Wvt  = (bf16*)(ws + 12 * MB);
  bf16*  Wot  = (bf16*)(ws + 14 * MB);
  bf16*  Wht  = (bf16*)(ws + 16 * MB);
  bf16*  Wyt  = (bf16*)(ws + 18 * MB);
  bf16*  qb   = (bf16*)(ws + 20 * MB);  // 8 MB
  bf16*  kb   = (bf16*)(ws + 28 * MB);  // 8 MB
  bf16*  vb   = (bf16*)(ws + 36 * MB);  // 8 MB
  bf16*  ctxb = (bf16*)(ws + 44 * MB);  // 8 MB
  float* hf   = (float*)(ws + 52 * MB); // 16 MB : h fp32 (residual source)
  bf16*  hb   = (bf16*)(ws + 68 * MB);  // 8 MB
  bf16*  ab   = (bf16*)(ws + 76 * MB);  // 8 MB  (84 MB total)

  const int nX = Mrows * Dc;
  cvt_kernel<<<nX / 256, 256, 0, stream>>>(x, xb, nX);

  const dim3 tb(32, 8), tg(Dc / 32, Dc / 32);
  wtrans_kernel<<<tg, tb, 0, stream>>>(Wq, Wqt);
  wtrans_kernel<<<tg, tb, 0, stream>>>(Wk, Wkt);
  wtrans_kernel<<<tg, tb, 0, stream>>>(Wv, Wvt);
  wtrans_kernel<<<tg, tb, 0, stream>>>(Wo, Wot);
  wtrans_kernel<<<tg, tb, 0, stream>>>(Wh, Wht);
  wtrans_kernel<<<tg, tb, 0, stream>>>(Wy, Wyt);

  const dim3 gb(Dc / 128, Mrows / 128); // (8, 32)
  // Q/K/V projections -> bf16
  gemm_bf16_kernel<EPI_NONE, false, true><<<gb, 256, 0, stream>>>(
      xb, Wqt, bq, nullptr, nullptr, qb, Mrows, Dc, Dc);
  gemm_bf16_kernel<EPI_NONE, false, true><<<gb, 256, 0, stream>>>(
      xb, Wkt, bk, nullptr, nullptr, kb, Mrows, Dc, Dc);
  gemm_bf16_kernel<EPI_NONE, false, true><<<gb, 256, 0, stream>>>(
      xb, Wvt, bv, nullptr, nullptr, vb, Mrows, Dc, Dc);

  // Attention
  flash_attn_kernel<<<dim3(Bc * Hc, Tc / 64), 128, 0, stream>>>(qb, kb, vb, ctxb);

  // h = ctx@Wo + bo + x  (fp32 + bf16 copies)
  gemm_bf16_kernel<EPI_RES, true, true><<<gb, 256, 0, stream>>>(
      ctxb, Wot, bo, x, hf, hb, Mrows, Dc, Dc);
  // a = leaky_relu(h@Wh + bh) -> bf16
  gemm_bf16_kernel<EPI_LEAKY, false, true><<<gb, 256, 0, stream>>>(
      hb, Wht, bh, nullptr, nullptr, ab, Mrows, Dc, Dc);
  // y = a@Wy + by + h -> d_out (fp32)
  gemm_bf16_kernel<EPI_RES, true, false><<<gb, 256, 0, stream>>>(
      ab, Wyt, by, hf, (float*)d_out, nullptr, Mrows, Dc, Dc);
}
